// HyperLuongDecoder_78486232367668
// MI455X (gfx1250) — matrix operand order repaired and checked by reading, not compiled, not run on hardware
//
#include <hip/hip_runtime.h>
#include <hip/hip_bf16.h>
#include <math.h>

#define T_LEN 32
#define BATCH 64
#define SRC   64
#define EMB   512
#define HID   1024
#define VOC   32000
#define LEMB  32

#define AS1 __attribute__((address_space(1)))

typedef __attribute__((ext_vector_type(16))) __bf16 v16bf;
typedef __attribute__((ext_vector_type(8)))  __bf16 v8bf;
typedef __attribute__((ext_vector_type(4)))  __bf16 v4bf;
typedef __attribute__((ext_vector_type(8)))  float  v8f;

__device__ __forceinline__ float sigmoidf_(float x) { return 1.0f / (1.0f + __expf(-x)); }

// ---------------------------------------------------------------------------
// WMMA fragment loaders (bf16, 16x16x32), contiguous-K order, same for A and B.
// p is a per-lane GLOBAL (addrspace 1) row pointer so the loads lower to
// global_load_b128 (LOADcnt only) instead of flat_load_b128 (LOADcnt+DScnt).
// eoff is a compile-time element offset folded into the load's imm offset.
// A frag: elems 0..15 = contiguous K[(lane>>4)*16..+16).
// B frag: elems 0..15 = contiguous K[(lane>>4)*16..+16).
// ---------------------------------------------------------------------------
__device__ __forceinline__ v16bf load_frag_a(const AS1 __bf16* p, int eoff) {
  v8bf c0 = *(const AS1 v8bf*)(p + eoff);
  v8bf c1 = *(const AS1 v8bf*)(p + eoff + 8);
  v16bf f;
#pragma unroll
  for (int i = 0; i < 8; ++i) { f[i] = c0[i]; f[i + 8] = c1[i]; }
  return f;
}

__device__ __forceinline__ v16bf load_frag_b(const AS1 __bf16* p, int eoff) {
  v8bf c0 = *(const AS1 v8bf*)(p + eoff);
  v8bf c1 = *(const AS1 v8bf*)(p + eoff + 8);
  v16bf f;
#pragma unroll
  for (int i = 0; i < 8; ++i) { f[i] = c0[i]; f[i + 8] = c1[i]; }
  return f;
}

// ---------------------------------------------------------------------------
// Generic bf16 WMMA GEMM: C[M,N] = A[M,K] * Bt[N,K]^T (+bias) (+tanh)
// Block = 8 waves. Wave tile = (MW*16) x 64; block tile = (MW*128) x 64.
// K-loop unrolled x2 with ping-pong fragment sets: compute set -> reload set,
// so loads of K+64 fly while the other set's WMMAs execute; no register
// rotation copies. Per-lane global row pointers advance once per 64-K iter.
// act: 0 = none, 1 = tanh.  outF (f32) and/or outB (bf16) destinations.
// M multiple of 16, N multiple of 64, K multiple of 64.
// ---------------------------------------------------------------------------
template <int MW>
__global__ __launch_bounds__(256)
void gemm_bf16_wmma(const __bf16* __restrict__ A, const __bf16* __restrict__ Bt,
                    const float* __restrict__ bias, float* __restrict__ outF,
                    __bf16* __restrict__ outB, int M, int N, int K, int act) {
  const int lane = threadIdx.x & 31;
  const int wave = threadIdx.x >> 5;
  const int mTile = blockIdx.y * (MW * 128) + wave * (MW * 16);
  if (mTile >= M) return;                       // wave-uniform; EXEC stays all-ones
  const int nTile0 = blockIdx.x * 64;

  // per-lane fragment row pointers, explicitly global address space
  const AS1 __bf16* pa[MW];
#pragma unroll
  for (int i = 0; i < MW; ++i)
    pa[i] = (const AS1 __bf16*)(A + (size_t)(mTile + 16 * i + (lane & 15)) * K +
                                ((lane >> 4) << 4));
  const AS1 __bf16* pb[4];
#pragma unroll
  for (int j = 0; j < 4; ++j)
    pb[j] = (const AS1 __bf16*)(Bt + (size_t)(nTile0 + 16 * j + (lane & 15)) * K +
                                ((lane >> 4) << 4));

  v8f acc[MW][4] = {};
  v16bf aA[MW], bA[4], aB[MW], bB[4];
#pragma unroll
  for (int i = 0; i < MW; ++i) aA[i] = load_frag_a(pa[i], 0);
#pragma unroll
  for (int j = 0; j < 4; ++j)  bA[j] = load_frag_b(pb[j], 0);
#pragma unroll
  for (int i = 0; i < MW; ++i) aB[i] = load_frag_a(pa[i], 32);
#pragma unroll
  for (int j = 0; j < 4; ++j)  bB[j] = load_frag_b(pb[j], 32);
#pragma unroll
  for (int i = 0; i < MW; ++i) pa[i] += 64;
#pragma unroll
  for (int j = 0; j < 4; ++j)  pb[j] += 64;

  for (int k0 = 64; k0 < K; k0 += 64) {
    // ---- consume set A (k0-64), refill with k0 ----
#pragma unroll
    for (int i = 0; i < MW; ++i)
#pragma unroll
      for (int j = 0; j < 4; ++j)
        acc[i][j] = __builtin_amdgcn_wmma_f32_16x16x32_bf16(false, aA[i], false, bA[j],
                                                            (short)0, acc[i][j], false, false);
#pragma unroll
    for (int i = 0; i < MW; ++i) aA[i] = load_frag_a(pa[i], 0);
#pragma unroll
    for (int j = 0; j < 4; ++j)  bA[j] = load_frag_b(pb[j], 0);
    // ---- consume set B (k0-32), refill with k0+32 ----
#pragma unroll
    for (int i = 0; i < MW; ++i)
#pragma unroll
      for (int j = 0; j < 4; ++j)
        acc[i][j] = __builtin_amdgcn_wmma_f32_16x16x32_bf16(false, aB[i], false, bB[j],
                                                            (short)0, acc[i][j], false, false);
#pragma unroll
    for (int i = 0; i < MW; ++i) aB[i] = load_frag_a(pa[i], 32);
#pragma unroll
    for (int j = 0; j < 4; ++j)  bB[j] = load_frag_b(pb[j], 32);
#pragma unroll
    for (int i = 0; i < MW; ++i) pa[i] += 64;
#pragma unroll
    for (int j = 0; j < 4; ++j)  pb[j] += 64;
  }
  // tail: both sets hold the last 64 of K
#pragma unroll
  for (int i = 0; i < MW; ++i)
#pragma unroll
    for (int j = 0; j < 4; ++j)
      acc[i][j] = __builtin_amdgcn_wmma_f32_16x16x32_bf16(false, aA[i], false, bA[j],
                                                          (short)0, acc[i][j], false, false);
#pragma unroll
  for (int i = 0; i < MW; ++i)
#pragma unroll
    for (int j = 0; j < 4; ++j)
      acc[i][j] = __builtin_amdgcn_wmma_f32_16x16x32_bf16(false, aB[i], false, bB[j],
                                                          (short)0, acc[i][j], false, false);

  // D layout: lanes 0-15 -> N=lane, M=tile+r ; lanes 16-31 -> N=lane-16, M=tile+8+r
#pragma unroll
  for (int i = 0; i < MW; ++i) {
    const int mBase = mTile + 16 * i + ((lane >> 4) << 3);
#pragma unroll
    for (int j = 0; j < 4; ++j) {
      const int n = nTile0 + j * 16 + (lane & 15);
      const float bv = bias ? bias[n] : 0.0f;
#pragma unroll
      for (int r = 0; r < 8; ++r) {
        float v = acc[i][j][r] + bv;
        if (act == 1) v = tanhf(v);
        const size_t off = (size_t)(mBase + r) * N + n;
        if (outF) outF[off] = v;
        if (outB) outB[off] = (__bf16)v;
      }
    }
  }
}

// ---------------------------------------------------------------------------
// Hypernetwork: out_bf16[i] = dot(W[i,:32], lang_embed[lang_idx]) + b[i]
// Each thread streams one 128B row; prefetch 256 rows (32 KB) ahead.
// ---------------------------------------------------------------------------
__global__ __launch_bounds__(256)
void hyper_matvec(const float* __restrict__ W, const float* __restrict__ bvec,
                  const float* __restrict__ lang_emb, const int* __restrict__ lidx,
                  __bf16* __restrict__ outp, long n) {
  long i = (long)blockIdx.x * blockDim.x + threadIdx.x;
  if (i >= n) return;
  if (i + 256 < n) __builtin_prefetch(W + (size_t)(i + 256) * LEMB, 0, 1);
  const float* el = lang_emb + (size_t)lidx[0] * LEMB;
  const float4* row = (const float4*)(W + (size_t)i * LEMB);
  float s = bvec[i];
#pragma unroll
  for (int q = 0; q < 8; ++q) {
    float4 w4 = row[q];
    s += w4.x * el[4 * q] + w4.y * el[4 * q + 1] + w4.z * el[4 * q + 2] + w4.w * el[4 * q + 3];
  }
  outp[i] = (__bf16)s;
}

__global__ __launch_bounds__(256)
void hyper_bias(const float* __restrict__ W_bih, const float* __restrict__ b_bih,
                const float* __restrict__ W_bhh, const float* __restrict__ b_bhh,
                const float* __restrict__ lang_emb, const int* __restrict__ lidx,
                float* __restrict__ outp) {
  int i = blockIdx.x * blockDim.x + threadIdx.x;
  if (i >= 4 * HID) return;
  const float* el = lang_emb + (size_t)lidx[0] * LEMB;
  float s = b_bih[i] + b_bhh[i];
#pragma unroll
  for (int j = 0; j < LEMB; ++j)
    s += (W_bih[(size_t)i * LEMB + j] + W_bhh[(size_t)i * LEMB + j]) * el[j];
  outp[i] = s;
}

// ---------------------------------------------------------------------------
// Embedding gather, vectorized 4-wide (E = 512 is a multiple of 4).
// ---------------------------------------------------------------------------
__global__ __launch_bounds__(256)
void embed_gather(const int* __restrict__ x, const float* __restrict__ vocab,
                  __bf16* __restrict__ emb) {
  long i4 = (long)blockIdx.x * blockDim.x + threadIdx.x;   // index in float4 units
  const long n4 = (long)T_LEN * BATCH * EMB / 4;
  if (i4 >= n4) return;
  long tb = i4 / (EMB / 4);
  int e4 = (int)(i4 % (EMB / 4));
  float4 v = ((const float4*)(vocab + (size_t)x[tb] * EMB))[e4];
  v4bf o;
  o[0] = (__bf16)v.x; o[1] = (__bf16)v.y; o[2] = (__bf16)v.z; o[3] = (__bf16)v.w;
  *(v4bf*)(emb + i4 * 4) = o;
}

// f32 -> bf16 stream, vectorized 4-wide (all sizes used are multiples of 4).
__global__ __launch_bounds__(256)
void f2bf4(const float4* __restrict__ in, __bf16* __restrict__ outp, long n4) {
  long i = (long)blockIdx.x * blockDim.x + threadIdx.x;
  if (i >= n4) return;
  float4 v = in[i];
  v4bf o;
  o[0] = (__bf16)v.x; o[1] = (__bf16)v.y; o[2] = (__bf16)v.z; o[3] = (__bf16)v.w;
  *(v4bf*)(outp + i * 4) = o;
}

// ---------------------------------------------------------------------------
// LSTM pointwise: gates = gates_h + gates_x + bsum; update c,h; emit bf16 h
// into concat buffer (second half of the 2H row) for the c2h GEMM.
// ---------------------------------------------------------------------------
__global__ __launch_bounds__(256)
void lstm_pointwise(const float* __restrict__ gates_h, const float* __restrict__ gates_x_t,
                    const float* __restrict__ bsum, float* __restrict__ c,
                    float* __restrict__ h_f32, __bf16* __restrict__ h_bf,
                    float* __restrict__ rnn_out_t, __bf16* __restrict__ concat_t) {
  int idx = blockIdx.x * blockDim.x + threadIdx.x;
  if (idx >= BATCH * HID) return;
  int b = idx / HID, j = idx % HID;
  size_t base = (size_t)b * 4 * HID + j;
  float ig = gates_h[base]           + gates_x_t[base]           + bsum[j];
  float fg = gates_h[base + HID]     + gates_x_t[base + HID]     + bsum[j + HID];
  float gg = gates_h[base + 2 * HID] + gates_x_t[base + 2 * HID] + bsum[j + 2 * HID];
  float og = gates_h[base + 3 * HID] + gates_x_t[base + 3 * HID] + bsum[j + 3 * HID];
  float cc = sigmoidf_(fg) * c[idx] + sigmoidf_(ig) * tanhf(gg);
  float hh = sigmoidf_(og) * tanhf(cc);
  c[idx] = cc;
  h_f32[idx] = hh;
  h_bf[idx] = (__bf16)hh;
  rnn_out_t[idx] = hh;
  concat_t[(size_t)b * 2 * HID + HID + j] = (__bf16)hh;
}

// ---------------------------------------------------------------------------
// Luong attention: one block per (t,b). Scores via wave-strided dots + shuffle
// reduce; masked softmax; context written (bf16) to first half of concat row.
// ---------------------------------------------------------------------------
__global__ __launch_bounds__(256)
void attention_kernel(const float* __restrict__ rnn_out, const float* __restrict__ src_enc,
                      const int* __restrict__ src_len, __bf16* __restrict__ concat) {
  const int tb = blockIdx.x;
  const int b = tb % BATCH;
  __shared__ float sc[SRC];
  __shared__ float invsum;
  const int wave = threadIdx.x >> 5, lane = threadIdx.x & 31;
  const float* q = rnn_out + (size_t)tb * HID;

  for (int s = wave; s < SRC; s += 8) {
    const float* kv = src_enc + ((size_t)s * BATCH + b) * HID;
    float sum = 0.f;
    for (int j = lane; j < HID; j += 32) sum += q[j] * kv[j];
#pragma unroll
    for (int off = 16; off > 0; off >>= 1) sum += __shfl_down(sum, off, 32);
    if (lane == 0) sc[s] = sum;
  }
  __syncthreads();

  if (threadIdx.x == 0) {
    const int L = src_len[b];
    float m = -3.0e38f;
    for (int s = 0; s < SRC; ++s) { if (s >= L) sc[s] = -1.0e30f; if (sc[s] > m) m = sc[s]; }
    float su = 0.f;
    for (int s = 0; s < SRC; ++s) { float e = __expf(sc[s] - m); sc[s] = e; su += e; }
    invsum = 1.0f / su;
  }
  __syncthreads();

  const float inv = invsum;
  for (int j = threadIdx.x; j < HID; j += 256) {
    float accv = 0.f;
    for (int s = 0; s < SRC; ++s)
      accv += sc[s] * src_enc[((size_t)s * BATCH + b) * HID + j];
    concat[(size_t)tb * (2 * HID) + j] = (__bf16)(accv * inv);
  }
}

// ---------------------------------------------------------------------------
extern "C" void kernel_launch(void* const* d_in, const int* in_sizes, int n_in,
                              void* d_out, int out_size, void* d_ws, size_t ws_size,
                              hipStream_t stream) {
  (void)in_sizes; (void)n_in; (void)out_size; (void)ws_size;

  const int*   x        = (const int*)d_in[0];
  const int*   src_len  = (const int*)d_in[1];
  const int*   lang_idx = (const int*)d_in[2];
  const float* h0       = (const float*)d_in[3];
  const float* c0       = (const float*)d_in[4];
  const float* src_enc  = (const float*)d_in[5];
  const float* vocab    = (const float*)d_in[6];
  const float* lang_emb = (const float*)d_in[7];
  const float* W_wih = (const float*)d_in[8];   const float* b_wih = (const float*)d_in[9];
  const float* W_whh = (const float*)d_in[10];  const float* b_whh = (const float*)d_in[11];
  const float* W_bih = (const float*)d_in[12];  const float* b_bih = (const float*)d_in[13];
  const float* W_bhh = (const float*)d_in[14];  const float* b_bhh = (const float*)d_in[15];
  const float* c2h_W = (const float*)d_in[16];  const float* c2h_b = (const float*)d_in[17];
  const float* h2o_W = (const float*)d_in[18];  const float* h2o_b = (const float*)d_in[19];
  float* out = (float*)d_out;

  // ---- carve workspace (256B aligned) ----
  char* p = (char*)d_ws;
  auto carve = [&](size_t bytes) -> void* {
    void* r = (void*)p;
    p += (bytes + 255) & ~(size_t)255;
    return r;
  };
  __bf16* w_ih_bf  = (__bf16*)carve((size_t)4 * HID * EMB * 2);       // (4H,E)
  __bf16* w_hh_bf  = (__bf16*)carve((size_t)4 * HID * HID * 2);       // (4H,H)
  float*  bsum     = (float*)carve((size_t)4 * HID * 4);              // (4H)
  __bf16* emb_bf   = (__bf16*)carve((size_t)T_LEN * BATCH * EMB * 2); // (TB,E)
  float*  gates_x  = (float*)carve((size_t)T_LEN * BATCH * 4 * HID * 4);
  float*  gates_t  = (float*)carve((size_t)BATCH * 4 * HID * 4);
  __bf16* h_bf     = (__bf16*)carve((size_t)BATCH * HID * 2);
  float*  h_f32    = (float*)carve((size_t)BATCH * HID * 4);
  float*  c_f32    = (float*)carve((size_t)BATCH * HID * 4);
  float*  rnn_out  = (float*)carve((size_t)T_LEN * BATCH * HID * 4);
  __bf16* concat_bf = (__bf16*)carve((size_t)T_LEN * BATCH * 2 * HID * 2);
  __bf16* attnh_bf = (__bf16*)carve((size_t)T_LEN * BATCH * HID * 2);
  __bf16* c2hW_bf  = (__bf16*)carve((size_t)HID * 2 * HID * 2);
  __bf16* h2oW_bf  = (__bf16*)carve((size_t)VOC * HID * 2);

  const dim3 blk(256);

  // 1. hypernetwork weights (memory-bound streaming, ~800 MB)
  {
    long n1 = (long)4 * HID * EMB;
    hyper_matvec<<<(unsigned)((n1 + 255) / 256), blk, 0, stream>>>(W_wih, b_wih, lang_emb, lang_idx, w_ih_bf, n1);
    long n2 = (long)4 * HID * HID;
    hyper_matvec<<<(unsigned)((n2 + 255) / 256), blk, 0, stream>>>(W_whh, b_whh, lang_emb, lang_idx, w_hh_bf, n2);
    hyper_bias<<<(4 * HID + 255) / 256, blk, 0, stream>>>(W_bih, b_bih, W_bhh, b_bhh, lang_emb, lang_idx, bsum);
  }

  // 2. weight / state conversions (4-wide vectorized streams)
  {
    long n4 = (long)T_LEN * BATCH * EMB / 4;
    embed_gather<<<(unsigned)((n4 + 255) / 256), blk, 0, stream>>>(x, vocab, emb_bf);
    long nh4 = (long)BATCH * HID / 4;
    f2bf4<<<(unsigned)((nh4 + 255) / 256), blk, 0, stream>>>((const float4*)h0, h_bf, nh4);
    hipMemcpyAsync(c_f32, c0, (size_t)BATCH * HID * sizeof(float), hipMemcpyDeviceToDevice, stream);
    long nc4 = (long)HID * 2 * HID / 4;
    f2bf4<<<(unsigned)((nc4 + 255) / 256), blk, 0, stream>>>((const float4*)c2h_W, c2hW_bf, nc4);
    long no4 = (long)VOC * HID / 4;
    f2bf4<<<(unsigned)((no4 + 255) / 256), blk, 0, stream>>>((const float4*)h2o_W, h2oW_bf, no4);
  }

  // 3. precompute input-gate GEMM for all timesteps: (2048,512)x(512,4096)
  gemm_bf16_wmma<2><<<dim3(4 * HID / 64, (T_LEN * BATCH) / 256), blk, 0, stream>>>(
      emb_bf, w_ih_bf, nullptr, gates_x, nullptr, T_LEN * BATCH, 4 * HID, EMB, 0);

  // 4. sequential LSTM: per step (64,1024)x(1024,4096) WMMA + pointwise gates
  //    MW=1 keeps 4 active waves/block (256 active waves) for the small-M GEMM.
  for (int t = 0; t < T_LEN; ++t) {
    gemm_bf16_wmma<1><<<dim3(4 * HID / 64, 1), blk, 0, stream>>>(
        h_bf, w_hh_bf, nullptr, gates_t, nullptr, BATCH, 4 * HID, HID, 0);
    lstm_pointwise<<<(BATCH * HID + 255) / 256, blk, 0, stream>>>(
        gates_t, gates_x + (size_t)t * BATCH * 4 * HID, bsum, c_f32, h_f32, h_bf,
        rnn_out + (size_t)t * BATCH * HID, concat_bf + (size_t)t * BATCH * 2 * HID);
  }

  // 5. attention -> context half of concat
  attention_kernel<<<T_LEN * BATCH, blk, 0, stream>>>(rnn_out, src_enc, src_len, concat_bf);

  // 6. attn_hidden = tanh(concat @ c2h_W^T + b): (2048,2048)x(2048,1024), bf16 out
  gemm_bf16_wmma<2><<<dim3(HID / 64, (T_LEN * BATCH) / 256), blk, 0, stream>>>(
      concat_bf, c2hW_bf, c2h_b, nullptr, attnh_bf, T_LEN * BATCH, HID, 2 * HID, 1);

  // 7. output projection: (2048,1024)x(1024,32000) + bias -> d_out (f32)
  gemm_bf16_wmma<2><<<dim3(VOC / 64, (T_LEN * BATCH) / 256), blk, 0, stream>>>(
      attnh_bf, h2oW_bf, h2o_b, out, nullptr, T_LEN * BATCH, VOC, HID, 0);

  // 8. hN, cN tails
  hipMemcpyAsync(out + (size_t)T_LEN * BATCH * VOC, h_f32,
                 (size_t)BATCH * HID * sizeof(float), hipMemcpyDeviceToDevice, stream);
  hipMemcpyAsync(out + (size_t)T_LEN * BATCH * VOC + (size_t)BATCH * HID, c_f32,
                 (size_t)BATCH * HID * sizeof(float), hipMemcpyDeviceToDevice, stream);
}
